// DiffAttnV2Like_40699110097640
// MI455X (gfx1250) — matrix operand order repaired and checked by reading, not compiled, not run on hardware
//
#include <hip/hip_runtime.h>
#include <stdint.h>

// Problem constants (fixed by the reference).
#define Bb   2
#define Tt   2048
#define Dd   2048
#define Hh   16
#define DHd  128
#define Mm   (Bb*Tt)          // 4096 rows total

typedef __bf16 bf16x16 __attribute__((ext_vector_type(16)));
typedef float  f32x8   __attribute__((ext_vector_type(8)));
typedef int    i32x4   __attribute__((ext_vector_type(4)));

union FragB16 { bf16x16 v; uint4 q[2]; };

__device__ __forceinline__ unsigned short f2bf(float f) {
  unsigned u = __float_as_uint(f);
  u += 0x7FFFu + ((u >> 16) & 1u);        // round-to-nearest-even
  return (unsigned short)(u >> 16);
}

__device__ __forceinline__ float fast_exp2(float x) {
  return __builtin_amdgcn_exp2f(x);       // v_exp_f32 (TRANS op, co-executes w/ WMMA)
}

// ---------------------------------------------------------------------------
// CDNA5 async global->LDS copy (ASYNCcnt path). 16B per lane per call.
// Builtin signature (from clang diagnostic): param0 = AS1 int4* (non-const).
// Falls back to a synchronous VGPR round-trip if the builtin is absent.
// ---------------------------------------------------------------------------
#if __has_builtin(__builtin_amdgcn_global_load_async_to_lds_b128)
#define HAS_ASYNC_LDS 1
#else
#define HAS_ASYNC_LDS 0
#endif

__device__ __forceinline__ void cp_async_b128(void* lds, const void* g) {
#if HAS_ASYNC_LDS
  __builtin_amdgcn_global_load_async_to_lds_b128(
      (__attribute__((address_space(1))) i32x4*)g,
      (__attribute__((address_space(3))) i32x4*)lds, 0, 0);
#else
  *(uint4*)lds = *(const uint4*)g;
#endif
}

__device__ __forceinline__ void cp_async_wait() {
#if HAS_ASYNC_LDS
  asm volatile("s_wait_asynccnt 0x0" ::: "memory");
#endif
}

// ---------------------------------------------------------------------------
// f32 -> bf16 conversion, 2 elems/thread, packed u32 stores.
// ---------------------------------------------------------------------------
__global__ void cvt_f32_bf16(const float* __restrict__ s, unsigned* __restrict__ d, int n2) {
  int i = blockIdx.x * blockDim.x + threadIdx.x;
  if (i < n2) {
    float2 f = ((const float2*)s)[i];
    d[i] = (unsigned)f2bf(f.x) | ((unsigned)f2bf(f.y) << 16);
  }
}

// ---------------------------------------------------------------------------
// bf16 WMMA GEMM: C[M,N] = A[M,K] * B[K,N], A/B bf16 row-major.
// 128x128 block tile, 256 threads = 8 waves in a 4(M) x 2(N) grid,
// each wave owns a 32x64 tile = 2x4 accumulators. K-step 32.
// Software-pipelined: double-buffered LDS, tile t+1 staged (A via async
// global->LDS, B via transposing copy) while tile t is computed; one
// barrier per iteration, preceded by s_wait_asynccnt.
// ---------------------------------------------------------------------------
template<bool OUT_BF16>
__global__ __launch_bounds__(256) void gemm_bf16(
    const unsigned short* __restrict__ A, const unsigned short* __restrict__ Bw,
    void* __restrict__ Cout, int Mdim, int Ndim, int Kdim)
{
  __shared__ unsigned short Ash[2][128 * 32];
  __shared__ unsigned short Bsh[2][128 * 32];  // transposed: [n][k]

  const int tid = threadIdx.x, lane = tid & 31, w = tid >> 5;
  const int l15 = lane & 15, hf = lane >> 4;
  const int m0 = blockIdx.y * 128, n0 = blockIdx.x * 128;
  const int mw = (w >> 1) * 32, nw = (w & 1) * 64;

  auto stage = [&](int buf, int k0) {
    // A tile 128x32: async 16B chunks straight into LDS (no VGPR round-trip).
#pragma unroll
    for (int c = tid; c < 512; c += 256) {
      int r = c >> 2, c4 = c & 3;
      cp_async_b128(&Ash[buf][r * 32 + c4 * 8],
                    &A[(size_t)(m0 + r) * Kdim + k0 + c4 * 8]);
    }
    // B tile 32x128 transposed: coalesced u32 reads along N, ushort writes.
#pragma unroll
    for (int c = tid; c < 2048; c += 256) {
      int kr = c >> 6, nc2 = (c & 63) * 2;
      unsigned v = *(const unsigned*)&Bw[(size_t)(k0 + kr) * Ndim + n0 + nc2];
      Bsh[buf][(nc2    ) * 32 + kr] = (unsigned short)(v & 0xFFFFu);
      Bsh[buf][(nc2 + 1) * 32 + kr] = (unsigned short)(v >> 16);
    }
  };

  f32x8 acc[2][4];
#pragma unroll
  for (int i = 0; i < 2; ++i)
#pragma unroll
    for (int j = 0; j < 4; ++j) acc[i][j] = (f32x8)0.f;

  stage(0, 0);
  cp_async_wait();
  __syncthreads();

  for (int k0 = 0; k0 < Kdim; k0 += 32) {
    const int cur = (k0 >> 5) & 1;
    if (k0 + 32 < Kdim) stage(cur ^ 1, k0 + 32);
    if (k0 + 64 < Kdim)   // pull k+2 weight rows toward L2/L0 (global_prefetch)
      __builtin_prefetch(&Bw[(size_t)(k0 + 64 + (tid >> 3)) * Ndim + n0 + (tid & 7) * 16], 0, 1);

    FragB16 af[2], bf[4];
#pragma unroll
    for (int fm = 0; fm < 2; ++fm) {
      int row = mw + fm * 16 + l15;
      af[fm].q[0] = *(const uint4*)&Ash[cur][row * 32 + hf * 8];
      af[fm].q[1] = *(const uint4*)&Ash[cur][row * 32 + hf * 8 + 16];
    }
#pragma unroll
    for (int fn = 0; fn < 4; ++fn) {
      int col = nw + fn * 16 + l15;
      bf[fn].q[0] = *(const uint4*)&Bsh[cur][col * 32 + hf * 8];
      bf[fn].q[1] = *(const uint4*)&Bsh[cur][col * 32 + hf * 8 + 16];
    }
#pragma unroll
    for (int fm = 0; fm < 2; ++fm)
#pragma unroll
      for (int fn = 0; fn < 4; ++fn)
        acc[fm][fn] = __builtin_amdgcn_wmma_f32_16x16x32_bf16(
            false, af[fm].v, false, bf[fn].v, (short)0, acc[fm][fn], false, false);

    cp_async_wait();   // next-buffer async copies complete
    __syncthreads();   // ... and everyone is done reading the current buffer
  }

  // Epilogue. C-layout: row = v + 8*half, col = lane&15.
#pragma unroll
  for (int fm = 0; fm < 2; ++fm)
#pragma unroll
    for (int fn = 0; fn < 4; ++fn)
#pragma unroll
      for (int v = 0; v < 8; ++v) {
        size_t row = (size_t)(m0 + mw + fm * 16 + v + hf * 8);
        size_t col = (size_t)(n0 + nw + fn * 16 + l15);
        if (OUT_BF16)
          ((unsigned short*)Cout)[row * Ndim + col] = f2bf(acc[fm][fn][v]);
        else
          ((float*)Cout)[row * Ndim + col] = acc[fm][fn][v];
      }
}

// ---------------------------------------------------------------------------
// lam[row,h] = sigmoid(sum_k x[row,k]*Wlam[k,h]).  1 block per row, 8 waves,
// each wave computes 2 heads via lane-strided dot + shfl_xor reduction.
// ---------------------------------------------------------------------------
__global__ __launch_bounds__(256) void lam_kernel(
    const float* __restrict__ x, const float* __restrict__ Wlam, float* __restrict__ lam)
{
  int row = blockIdx.x;
  int lane = threadIdx.x & 31, w = threadIdx.x >> 5;
  const float* xr = x + (size_t)row * Dd;
  int h0 = w * 2;
  float a0 = 0.f, a1 = 0.f;
  for (int k = lane; k < Dd; k += 32) {
    float xv = xr[k];
    a0 += xv * Wlam[k * Hh + h0];
    a1 += xv * Wlam[k * Hh + h0 + 1];
  }
#pragma unroll
  for (int m = 16; m >= 1; m >>= 1) { a0 += __shfl_xor(a0, m); a1 += __shfl_xor(a1, m); }
  if (lane == 0) {
    lam[row * Hh + h0]     = 1.f / (1.f + __expf(-a0));
    lam[row * Hh + h0 + 1] = 1.f / (1.f + __expf(-a1));
  }
}

// ---------------------------------------------------------------------------
// Fused differential flash attention + (y1 - lam*y2) combine.
// Block = (64 q-rows) x (output head h).  Waves 0-3: q1 vs kv-head h/2;
// waves 4-7: q2 vs kv-head 8+h/2 (interleaved repeat(k,2) pairing!).
// K tiles staged with async global->LDS; V tiles staged transposed.
// Online softmax over WMMA C-layout rows with wave32 shfl_xor (16-lane groups).
// P converted C-layout -> A-layout via per-wave LDS + s_wait_dscnt.
// ---------------------------------------------------------------------------
__global__ __launch_bounds__(256) void flash_diff(
    const unsigned short* __restrict__ Q1, const unsigned short* __restrict__ Q2,
    const unsigned short* __restrict__ Kp, const unsigned short* __restrict__ Vp,
    const float* __restrict__ lam, unsigned short* __restrict__ yc)
{
  __shared__ unsigned short Ksh [2 * 64 * 128];   // [set][key][d]
  __shared__ unsigned short Vtsh[2 * 128 * 64];   // [set][d][key]
  __shared__ unsigned short Psh [8][16 * 64];     // per-wave P round-trip
  __shared__ float          comb[64 * 128];       // lam*y2 hand-off

  const int tid = threadIdx.x, lane = tid & 31, w = tid >> 5;
  const int l15 = lane & 15, hf = lane >> 4;
  const int b = blockIdx.z, h = blockIdx.y, qblk = blockIdx.x;
  const int tq0 = qblk * 64;
  const int qsel = w >> 2;            // 0: q1-group, 1: q2-group
  const int wr = (w & 3) * 16;        // wave's row offset within the 64-row block
  const unsigned short* Qp = qsel ? Q2 : Q1;
  const int kvh0 = (h >> 1);          // kv head for q1
  const int kvh1 = 8 + (h >> 1);      // kv head for q2

  // Q fragments (A-layout), resident in VGPRs for the whole kernel.
  FragB16 qf[4];
  {
    size_t r = (size_t)(b * Tt + tq0 + wr + l15) * Dd + h * DHd;
#pragma unroll
    for (int kd = 0; kd < 4; ++kd) {
      qf[kd].q[0] = *(const uint4*)&Qp[r + kd * 32 + hf * 8];
      qf[kd].q[1] = *(const uint4*)&Qp[r + kd * 32 + hf * 8 + 16];
    }
  }

  f32x8 yacc[8];
#pragma unroll
  for (int i = 0; i < 8; ++i) yacc[i] = (f32x8)0.f;
  float mrow[8], lrow[8];
#pragma unroll
  for (int v = 0; v < 8; ++v) { mrow[v] = -1e30f; lrow[v] = 0.f; }

  const float F = 0.08838834764831845f * 1.4426950408889634f; // (1/sqrt(128))*log2(e)
  const int Kbase = qsel * (64 * 128);
  const int Vbase = qsel * (128 * 64);

  for (int kt = 0; kt <= qblk; ++kt) {
    const int kt0 = kt * 64;
    // Stage both K tile sets: 2 x 64x128 bf16, async 16B chunks.
#pragma unroll
    for (int c = tid; c < 2048; c += 256) {
      int s = c >> 10, r = (c >> 4) & 63, c16 = c & 15;
      int kvh = s ? kvh1 : kvh0;
      cp_async_b128(&Ksh[s * (64 * 128) + r * 128 + c16 * 8],
                    &Kp[(size_t)(b * Tt + kt0 + r) * Dd + kvh * DHd + c16 * 8]);
    }
    // Stage both V tile sets transposed: u32 reads (2 d-elems), ushort writes.
#pragma unroll
    for (int c = tid; c < 8192; c += 256) {
      int s = c >> 12, k = (c >> 6) & 63, d2 = (c & 63) * 2;
      int kvh = s ? kvh1 : kvh0;
      unsigned v = *(const unsigned*)&Vp[(size_t)(b * Tt + kt0 + k) * Dd + kvh * DHd + d2];
      Vtsh[s * (128 * 64) + (d2    ) * 64 + k] = (unsigned short)(v & 0xFFFFu);
      Vtsh[s * (128 * 64) + (d2 + 1) * 64 + k] = (unsigned short)(v >> 16);
    }
    cp_async_wait();
    __syncthreads();

    // S = Q * K^T over d=128 (4 k-steps), 4 column fragments (64 keys).
    f32x8 s[4];
#pragma unroll
    for (int fn = 0; fn < 4; ++fn) s[fn] = (f32x8)0.f;
#pragma unroll
    for (int fn = 0; fn < 4; ++fn)
#pragma unroll
      for (int kd = 0; kd < 4; ++kd) {
        FragB16 bk;
        const int base = Kbase + (fn * 16 + l15) * 128 + kd * 32 + hf * 8;
        bk.q[0] = *(const uint4*)&Ksh[base];
        bk.q[1] = *(const uint4*)&Ksh[base + 16];
        s[fn] = __builtin_amdgcn_wmma_f32_16x16x32_bf16(
            false, qf[kd].v, false, bk.v, (short)0, s[fn], false, false);
      }

    // Causal mask on the diagonal tile.
    if (kt == qblk) {
#pragma unroll
      for (int fn = 0; fn < 4; ++fn)
#pragma unroll
        for (int v = 0; v < 8; ++v)
          if (fn * 16 + l15 > wr + v + hf * 8) s[fn][v] = -1e30f;
    }

    // Online softmax: rows live across 16-lane half-wave groups.
#pragma unroll
    for (int v = 0; v < 8; ++v) {
      float tmax = fmaxf(fmaxf(s[0][v], s[1][v]), fmaxf(s[2][v], s[3][v]));
#pragma unroll
      for (int m = 8; m >= 1; m >>= 1) tmax = fmaxf(tmax, __shfl_xor(tmax, m));
      float nm = fmaxf(mrow[v], tmax);
      float corr = fast_exp2((mrow[v] - nm) * F);
      mrow[v] = nm;
      float psum = 0.f;
#pragma unroll
      for (int fn = 0; fn < 4; ++fn) {
        float p = fast_exp2((s[fn][v] - nm) * F);
        s[fn][v] = p;
        psum += p;
      }
#pragma unroll
      for (int m = 8; m >= 1; m >>= 1) psum += __shfl_xor(psum, m);
      lrow[v] = lrow[v] * corr + psum;
#pragma unroll
      for (int dn = 0; dn < 8; ++dn) yacc[dn][v] *= corr;
    }

    // P: C-layout -> per-wave LDS -> A-layout fragments.
#pragma unroll
    for (int fn = 0; fn < 4; ++fn)
#pragma unroll
      for (int v = 0; v < 8; ++v)
        Psh[w][(v + hf * 8) * 64 + fn * 16 + l15] = f2bf(s[fn][v]);
    asm volatile("s_wait_dscnt 0x0" ::: "memory");

    FragB16 pf[2];
#pragma unroll
    for (int fk = 0; fk < 2; ++fk) {
      pf[fk].q[0] = *(const uint4*)&Psh[w][l15 * 64 + fk * 32 + hf * 8];
      pf[fk].q[1] = *(const uint4*)&Psh[w][l15 * 64 + fk * 32 + hf * 8 + 16];
    }
    // y += P * V  (8 d-fragments x 2 k-steps).
#pragma unroll
    for (int dn = 0; dn < 8; ++dn)
#pragma unroll
      for (int fk = 0; fk < 2; ++fk) {
        FragB16 bv;
        const int base = Vbase + (dn * 16 + l15) * 64 + fk * 32 + hf * 8;
        bv.q[0] = *(const uint4*)&Vtsh[base];
        bv.q[1] = *(const uint4*)&Vtsh[base + 16];
        yacc[dn] = __builtin_amdgcn_wmma_f32_16x16x32_bf16(
            false, pf[fk].v, false, bv.v, (short)0, yacc[dn], false, false);
      }
    __syncthreads();   // protect Ksh/Vtsh before next tile
  }

  // Combine: waves 4-7 write lam*y2 (normalized) to LDS; waves 0-3 subtract.
  if (qsel == 1) {
#pragma unroll
    for (int v = 0; v < 8; ++v) {
      int rloc = wr + v + hf * 8;
      float lamv = lam[(size_t)(b * Tt + tq0 + rloc) * Hh + h];
      float g = lamv / lrow[v];
#pragma unroll
      for (int dn = 0; dn < 8; ++dn)
        comb[rloc * 128 + dn * 16 + l15] = yacc[dn][v] * g;
    }
  }
  __syncthreads();
  if (qsel == 0) {
#pragma unroll
    for (int v = 0; v < 8; ++v) {
      int rloc = wr + v + hf * 8;
      float inv = 1.f / lrow[v];
      size_t orow = (size_t)(b * Tt + tq0 + rloc) * Dd + h * DHd;
#pragma unroll
      for (int dn = 0; dn < 8; ++dn) {
        float val = yacc[dn][v] * inv - comb[rloc * 128 + dn * 16 + l15];
        yc[orow + dn * 16 + l15] = f2bf(val);
      }
    }
  }
}

// ---------------------------------------------------------------------------
extern "C" void kernel_launch(void* const* d_in, const int* in_sizes, int n_in,
                              void* d_out, int out_size, void* d_ws, size_t ws_size,
                              hipStream_t stream) {
  const float* x    = (const float*)d_in[0];
  const float* Wq1  = (const float*)d_in[1];
  const float* Wq2  = (const float*)d_in[2];
  const float* Wk   = (const float*)d_in[3];
  const float* Wv   = (const float*)d_in[4];
  const float* Wlam = (const float*)d_in[5];
  const float* Wo   = (const float*)d_in[6];

  unsigned char* ws = (unsigned char*)d_ws;
  size_t off = 0;
  auto alloc = [&](size_t bytes) -> void* {
    void* p = ws + off;
    off += (bytes + 255) & ~(size_t)255;
    return p;
  };
  unsigned short* xb   = (unsigned short*)alloc((size_t)Mm * Dd * 2);
  unsigned short* wq1b = (unsigned short*)alloc((size_t)Dd * Dd * 2);
  unsigned short* wq2b = (unsigned short*)alloc((size_t)Dd * Dd * 2);
  unsigned short* wkb  = (unsigned short*)alloc((size_t)Dd * Dd * 2);
  unsigned short* wvb  = (unsigned short*)alloc((size_t)Dd * Dd * 2);
  unsigned short* wob  = (unsigned short*)alloc((size_t)Dd * Dd * 2);
  unsigned short* q1   = (unsigned short*)alloc((size_t)Mm * Dd * 2);
  unsigned short* q2   = (unsigned short*)alloc((size_t)Mm * Dd * 2);
  unsigned short* kp   = (unsigned short*)alloc((size_t)Mm * Dd * 2);
  unsigned short* vp   = (unsigned short*)alloc((size_t)Mm * Dd * 2);
  float*          lamb = (float*)alloc((size_t)Mm * Hh * 4);
  unsigned short* yc   = (unsigned short*)alloc((size_t)Mm * Dd * 2);
  (void)ws_size; (void)n_in; (void)in_sizes; (void)out_size;

  // 1) f32 -> bf16 conversions.
  {
    int n2 = Mm * Dd / 2;
    cvt_f32_bf16<<<n2 / 256, 256, 0, stream>>>(x, (unsigned*)xb, n2);
    int w2 = Dd * Dd / 2;
    cvt_f32_bf16<<<w2 / 256, 256, 0, stream>>>(Wq1, (unsigned*)wq1b, w2);
    cvt_f32_bf16<<<w2 / 256, 256, 0, stream>>>(Wq2, (unsigned*)wq2b, w2);
    cvt_f32_bf16<<<w2 / 256, 256, 0, stream>>>(Wk,  (unsigned*)wkb,  w2);
    cvt_f32_bf16<<<w2 / 256, 256, 0, stream>>>(Wv,  (unsigned*)wvb,  w2);
    cvt_f32_bf16<<<w2 / 256, 256, 0, stream>>>(Wo,  (unsigned*)wob,  w2);
  }

  // 2) Projections: Q1/Q2/K/V = xb @ W*, bf16 out.
  dim3 gg(Dd / 128, Mm / 128);
  gemm_bf16<true><<<gg, 256, 0, stream>>>(xb, wq1b, q1, Mm, Dd, Dd);
  gemm_bf16<true><<<gg, 256, 0, stream>>>(xb, wq2b, q2, Mm, Dd, Dd);
  gemm_bf16<true><<<gg, 256, 0, stream>>>(xb, wkb,  kp, Mm, Dd, Dd);
  gemm_bf16<true><<<gg, 256, 0, stream>>>(xb, wvb,  vp, Mm, Dd, Dd);

  // 3) Gates.
  lam_kernel<<<Mm, 256, 0, stream>>>(x, Wlam, lamb);

  // 4) Fused differential flash attention -> yc (bf16, merged heads).
  flash_diff<<<dim3(Tt / 64, Hh, Bb), 256, 0, stream>>>(q1, q2, kp, vp, lamb, yc);

  // 5) Output projection: d_out(f32) = yc @ Wo.
  gemm_bf16<false><<<gg, 256, 0, stream>>>(yc, wob, (float*)d_out, Mm, Dd, Dd);
}
